// BehlerParrinelloNeuralNetwork_8675833938301
// MI455X (gfx1250) — matrix operand order repaired and checked by reading, not compiled, not run on hardware
//
#include <hip/hip_runtime.h>

typedef __attribute__((ext_vector_type(16))) __bf16 v16bf;
typedef __attribute__((ext_vector_type(8)))  __bf16 v8bf;
typedef __attribute__((ext_vector_type(8)))  float  v8f;

#define T_  4
#define N_  131072
#define F_  128
#define H1_ 500
#define H2_ 200
#define H3_ 100
#define P1  512   // H1 padded to mult of 32
#define P2  224   // H2 padded
#define P3  128   // H3 padded
#define M_  16384

#define WAVES 4
#define ATOMS (WAVES * 16)

// ---------------------------------------------------------------- WMMA helpers

__device__ __forceinline__ v8f wmma_bf16(v16bf a, v16bf b, v8f c) {
  // D = A(16x32) * B(32x16) + C, f32 accumulate
  return __builtin_amdgcn_wmma_f32_16x16x32_bf16(
      /*neg_a=*/false, a, /*neg_b=*/false, b,
      /*c_mod=*/(short)0, c, /*reuse_a=*/false, /*reuse_b=*/false);
}

__device__ __forceinline__ float relu1(float f) {
  return __builtin_amdgcn_fmed3f(f, 0.0f, __builtin_huge_valf());
}

// A fragment (16x32 bf16) from LDS at precomputed per-lane base + constant k0.
// aBase = tile + (lane&15)*ldk + ((lane&16)?8:0); k0 folds into ds imm offsets.
__device__ __forceinline__ v16bf loadA_lds(const __bf16* aBase, int k0) {
  v8bf lo = *(const v8bf*)(aBase + k0);        // K kk..kk+7
  v8bf hi = *(const v8bf*)(aBase + k0 + 16);   // K kk+16..kk+23
  v16bf a;
#pragma unroll
  for (int i = 0; i < 8; ++i) { a[i] = lo[i]; a[8 + i] = hi[i]; }
  return a;
}

// A fragment straight from global f32 (converted to bf16 in registers).
__device__ __forceinline__ v16bf loadA_gf32(const float* base, int ldk, int k0, int lane) {
  int m  = lane & 15;
  int kk = k0 + ((lane & 16) ? 8 : 0);
  const float* p = base + (size_t)m * ldk + kk;
  v16bf a;
#pragma unroll
  for (int i = 0; i < 8; ++i) { a[i] = (__bf16)p[i]; a[8 + i] = (__bf16)p[16 + i]; }
  return a;
}

// B fragment (32x16 bf16): uniform base pointer + 32-bit per-lane element offset.
// Encourages SADDR(SGPR64) + VGPR32-offset global_load selection.
__device__ __forceinline__ v16bf loadB(const __bf16* __restrict__ W, int off) {
  v8bf lo = *(const v8bf*)(W + off);        // K kk..kk+7, 16B aligned
  v8bf hi = *(const v8bf*)(W + off + 16);   // K kk+16..kk+23
  v16bf b;
#pragma unroll
  for (int i = 0; i < 8; ++i) { b[i] = lo[i]; b[8 + i] = hi[i]; }
  return b;
}

// Store D tile (optional ReLU) as bf16; q = tile + mb*ldn + col(lane), row offset
// v*ldn is compile-time -> folds into ds_store_b16 immediate offsets.
__device__ __forceinline__ void storeD(__bf16* q, int ldn, v8f c, bool relu) {
#pragma unroll
  for (int v = 0; v < 8; ++v) {
    float f = c[v];
    if (relu) f = relu1(f);
    q[v * ldn] = (__bf16)f;
  }
}

// ---------------------------------------------------------------- prep kernels

__global__ void cvt_pad_kernel(const float* __restrict__ src, __bf16* __restrict__ dst,
                               int Hout, int Kin, int Hpad, int Kpad) {
  size_t tot = (size_t)T_ * Hpad * Kpad;
  for (size_t i = (size_t)blockIdx.x * blockDim.x + threadIdx.x; i < tot;
       i += (size_t)gridDim.x * blockDim.x) {
    int k = (int)(i % Kpad);
    size_t r = i / Kpad;
    int n = (int)(r % Hpad);
    int t = (int)(r / Hpad);
    float v = (n < Hout && k < Kin) ? src[((size_t)t * Hout + n) * Kin + k] : 0.0f;
    dst[i] = (__bf16)v;
  }
}

__global__ void pad_bias_kernel(const float* __restrict__ src, float* __restrict__ dst,
                                int Hout, int Hpad) {
  int tot = T_ * Hpad;
  for (int i = blockIdx.x * blockDim.x + threadIdx.x; i < tot;
       i += gridDim.x * blockDim.x) {
    int n = i % Hpad;
    int t = i / Hpad;
    dst[i] = (n < Hout) ? src[t * Hout + n] : 0.0f;
  }
}

__global__ void init_e_kernel(float* __restrict__ out, const float* __restrict__ e0, int n) {
  int i = blockIdx.x * blockDim.x + threadIdx.x;
  if (i < n) out[i] = e0[i];
}

// ---------------------------------------------------------------- fused MLP

__global__ __launch_bounds__(WAVES * 32)
void bp_fused_kernel(const float* __restrict__ x, const int* __restrict__ ind,
                     float* __restrict__ e,
                     const __bf16* __restrict__ W1p, const __bf16* __restrict__ W2p,
                     const __bf16* __restrict__ W3p, const __bf16* __restrict__ W4p,
                     const float* __restrict__ b1p, const float* __restrict__ b2p,
                     const float* __restrict__ b3p, const float* __restrict__ b4p) {
  __shared__ alignas(16) __bf16 ldsA[WAVES * 16 * P1];  // h1, later reused for h3
  __shared__ alignas(16) __bf16 ldsB[WAVES * 16 * P2];  // h2

  const int lane = threadIdx.x & 31;
  const int wave = threadIdx.x >> 5;
  const int wgPerT = N_ / ATOMS;
  const int t = blockIdx.x / wgPerT;
  const int rowBase = (blockIdx.x % wgPerT) * ATOMS + wave * 16;

  __bf16* h1 = ldsA + wave * 16 * P1;
  __bf16* h2 = ldsB + wave * 16 * P2;
  __bf16* h3 = ldsA + wave * 16 * P1;  // h1 dead once layer 2 done

  const int nl   = lane & 15;
  const int hi16 = (lane >> 4) & 1;
  const int mb   = hi16 * 8;            // D-row base for this half-wave
  const int aSel = hi16 * 8;            // A fragment K sub-offset
  const int bSel = hi16 * 8;            // B fragment K sub-offset

  // ---- Layer 1: x(16x128) @ W1^T -> h1(16x500, pad 512), ReLU
  {
    const float* xt = x + ((size_t)t * N_ + rowBase) * F_;
    v16bf xa[4];
#pragma unroll
    for (int kc = 0; kc < 4; ++kc) xa[kc] = loadA_gf32(xt, F_, kc * 32, lane);

    const __bf16* W1t = W1p + (size_t)t * P1 * F_;
    const float*  b1t = b1p + t * P1;
    const int     bL1 = nl * F_ + bSel;     // per-lane B offset within a tile row-block
    __bf16*       s1  = h1 + mb * P1 + nl;  // per-lane store base
#pragma unroll 2
    for (int nt = 0; nt < P1 / 16; ++nt) {
      float bias = b1t[nt * 16 + nl];
      v8f c;
#pragma unroll
      for (int v = 0; v < 8; ++v) c[v] = bias;
#pragma unroll
      for (int kc = 0; kc < 4; ++kc) {
        v16bf b = loadB(W1t, nt * 16 * F_ + bL1 + kc * 32);
        c = wmma_bf16(xa[kc], b, c);
      }
      storeD(s1 + nt * 16, P1, c, true);
    }
  }

  // ---- Layer 2: h1(16x512) @ W2^T -> h2(16x200, pad 224), ReLU (2-wide N blocking)
  {
    const __bf16* W2t = W2p + (size_t)t * P2 * P1;
    const float*  b2t = b2p + t * P2;
    const int     bL2 = nl * P1 + bSel;
    const __bf16* a2  = h1 + nl * P1 + aSel;   // per-lane A base (const k0 -> ds imm)
    __bf16*       s2  = h2 + mb * P2 + nl;
#pragma unroll 1
    for (int np = 0; np < P2 / 32; ++np) {
      int nt0 = np * 2, nt1 = np * 2 + 1;
      float bias0 = b2t[nt0 * 16 + nl];
      float bias1 = b2t[nt1 * 16 + nl];
      v8f c0, c1;
#pragma unroll
      for (int v = 0; v < 8; ++v) { c0[v] = bias0; c1[v] = bias1; }
#pragma unroll
      for (int kc = 0; kc < P1 / 32; ++kc) {
        v16bf a   = loadA_lds(a2, kc * 32);
        v16bf bb0 = loadB(W2t, nt0 * 16 * P1 + bL2 + kc * 32);
        v16bf bb1 = loadB(W2t, nt1 * 16 * P1 + bL2 + kc * 32);
        c0 = wmma_bf16(a, bb0, c0);
        c1 = wmma_bf16(a, bb1, c1);
      }
      storeD(s2 + nt0 * 16, P2, c0, true);
      storeD(s2 + nt1 * 16, P2, c1, true);
    }
  }

  // ---- Layer 3: h2(16x224) @ W3^T -> h3(16x100, pad 128), ReLU (A hoisted)
  {
    const __bf16* a3 = h2 + nl * P2 + aSel;
    v16bf ha[P2 / 32];
#pragma unroll
    for (int kc = 0; kc < P2 / 32; ++kc) ha[kc] = loadA_lds(a3, kc * 32);

    const __bf16* W3t = W3p + (size_t)t * P3 * P2;
    const float*  b3t = b3p + t * P3;
    const int     bL3 = nl * P2 + bSel;
    __bf16*       s3  = h3 + mb * P3 + nl;
#pragma unroll 2
    for (int nt = 0; nt < P3 / 16; ++nt) {
      float bias = b3t[nt * 16 + nl];
      v8f c;
#pragma unroll
      for (int v = 0; v < 8; ++v) c[v] = bias;
#pragma unroll
      for (int kc = 0; kc < P2 / 32; ++kc) {
        v16bf b = loadB(W3t, nt * 16 * P2 + bL3 + kc * 32);
        c = wmma_bf16(ha[kc], b, c);
      }
      storeD(s3 + nt * 16, P3, c, true);
    }
  }

  // ---- Layer 4: h3(16x128) @ W4^T -> per-atom scalar; scatter-add into e[ind]
  {
    const __bf16* W4t = W4p + (size_t)t * 16 * P3;
    const __bf16* a4  = h3 + nl * P3 + aSel;
    const int     bL4 = nl * P3 + bSel;
    float bias4 = b4p[t * 16 + nl];
    v8f c;
#pragma unroll
    for (int v = 0; v < 8; ++v) c[v] = bias4;
#pragma unroll
    for (int kc = 0; kc < 4; ++kc) {
      v16bf a = loadA_lds(a4, kc * 32);
      v16bf b = loadB(W4t, bL4 + kc * 32);
      c = wmma_bf16(a, b, c);
    }
    if (nl == 0) {  // lanes 0 (rows 0-7) and 16 (rows 8-15) hold column N=0
      const int* indt = ind + (size_t)t * N_ + rowBase;
#pragma unroll
      for (int v = 0; v < 8; ++v) atomicAdd(&e[indt[mb + v]], c[v]);
    }
  }
}

// ---------------------------------------------------------------- launch

extern "C" void kernel_launch(void* const* d_in, const int* in_sizes, int n_in,
                              void* d_out, int out_size, void* d_ws, size_t ws_size,
                              hipStream_t stream) {
  const float* x   = (const float*)d_in[0];
  const int*   ind = (const int*)  d_in[1];
  const float* e0  = (const float*)d_in[2];
  const float* W1  = (const float*)d_in[3];
  const float* b1  = (const float*)d_in[4];
  const float* W2  = (const float*)d_in[5];
  const float* b2  = (const float*)d_in[6];
  const float* W3  = (const float*)d_in[7];
  const float* b3  = (const float*)d_in[8];
  const float* W4  = (const float*)d_in[9];
  const float* b4  = (const float*)d_in[10];
  float* out = (float*)d_out;

  // Workspace layout: padded bf16 weights (~1.69 MB) + padded f32 biases (~14 KB)
  __bf16* W1p = (__bf16*)d_ws;                       // [T][P1][F_]
  __bf16* W2p = W1p + (size_t)T_ * P1 * F_;          // [T][P2][P1]
  __bf16* W3p = W2p + (size_t)T_ * P2 * P1;          // [T][P3][P2]
  __bf16* W4p = W3p + (size_t)T_ * P3 * P2;          // [T][16][P3]
  float*  b1pd = (float*)(W4p + (size_t)T_ * 16 * P3);  // [T][P1]
  float*  b2pd = b1pd + T_ * P1;                        // [T][P2]
  float*  b3pd = b2pd + T_ * P2;                        // [T][P3]
  float*  b4pd = b3pd + T_ * P3;                        // [T][16]

  {
    size_t tot = (size_t)T_ * P1 * F_;
    cvt_pad_kernel<<<(int)((tot + 255) / 256), 256, 0, stream>>>(W1, W1p, H1_, F_, P1, F_);
  }
  {
    size_t tot = (size_t)T_ * P2 * P1;
    cvt_pad_kernel<<<(int)((tot + 255) / 256), 256, 0, stream>>>(W2, W2p, H2_, H1_, P2, P1);
  }
  {
    size_t tot = (size_t)T_ * P3 * P2;
    cvt_pad_kernel<<<(int)((tot + 255) / 256), 256, 0, stream>>>(W3, W3p, H3_, H2_, P3, P2);
  }
  {
    size_t tot = (size_t)T_ * 16 * P3;
    cvt_pad_kernel<<<(int)((tot + 255) / 256), 256, 0, stream>>>(W4, W4p, 1, H3_, 16, P3);
  }
  pad_bias_kernel<<<(T_ * P1 + 255) / 256, 256, 0, stream>>>(b1, b1pd, H1_, P1);
  pad_bias_kernel<<<(T_ * P2 + 255) / 256, 256, 0, stream>>>(b2, b2pd, H2_, P2);
  pad_bias_kernel<<<(T_ * P3 + 255) / 256, 256, 0, stream>>>(b3, b3pd, H3_, P3);
  pad_bias_kernel<<<(T_ * 16 + 255) / 256, 256, 0, stream>>>(b4, b4pd, 1, 16);

  init_e_kernel<<<(out_size + 255) / 256, 256, 0, stream>>>(out, e0, out_size);

  dim3 grid(T_ * (N_ / ATOMS));
  bp_fused_kernel<<<grid, WAVES * 32, 0, stream>>>(x, ind, out,
                                                   W1p, W2p, W3p, W4p,
                                                   b1pd, b2pd, b3pd, b4pd);
}
